// Conductor_89404039234004
// MI455X (gfx1250) — matrix operand order repaired and checked by reading, not compiled, not run on hardware
//
#include <hip/hip_runtime.h>

#define HDIM 1024
#define LNUM 2
#define TSTEPS 32
#define BSZ 512
#define GDIM 4096  // 4*H

typedef __attribute__((ext_vector_type(16))) __bf16 v16bf;
typedef __attribute__((ext_vector_type(8)))  float  v8f;
typedef __attribute__((ext_vector_type(4)))  __bf16 v4bf;

// ---------------------------------------------------------------------------
// WMMA helpers (CDNA5 gfx1250: V_WMMA_F32_16X16X32_BF16, wave32)
// ---------------------------------------------------------------------------
__device__ __forceinline__ v8f wmma_bf16(v16bf a, v16bf b, v8f c) {
  // (neg_a, A, neg_b, B, c_mod, C, reuse_a, reuse_b)
  return __builtin_amdgcn_wmma_f32_16x16x32_bf16(false, a, false, b, (short)0, c,
                                                 false, false);
}

union Frag16 { uint4 u[2]; v16bf v; };

// A-matrix 16x32 bf16 layout (ISA 7.12.2):
//   lanes 0-15 : row M=lane,    K = {0..7, 16..23}
//   lanes 16-31: row M=lane-16, K = {8..15, 24..31}
// rowk must already point at A[row] + k0 (contiguous along K).
__device__ __forceinline__ v16bf load_a_frag(const __bf16* rowk, int lane) {
  const int kb = (lane >> 4) << 3;  // 0 or 8
  Frag16 f;
  f.u[0] = *reinterpret_cast<const uint4*>(rowk + kb);
  f.u[1] = *reinterpret_cast<const uint4*>(rowk + 16 + kb);
  return f.v;
}

// B-matrix 32x16 bf16 (K x N), column n = lane&15 held per lane:
//   lanes 0-15 : K = {0..7, 16..23} ; lanes 16-31 : K = {8..15, 24..31}
// rowk points at W[n] + k0 (weights stored row-major [N][K], i.e. B transposed).
__device__ __forceinline__ v16bf load_b_frag(const __bf16* rowk, int lane) {
  const int kb = (lane >> 4) << 3;  // 0 or 8
  Frag16 f;
  f.u[0] = *reinterpret_cast<const uint4*>(rowk + kb);
  f.u[1] = *reinterpret_cast<const uint4*>(rowk + 16 + kb);
  return f.v;
}

__device__ __forceinline__ float sigm_f(float x) {
  return 1.0f / (1.0f + __expf(-x));
}
__device__ __forceinline__ float tanh_f(float x) {
  return 1.0f - 2.0f / (__expf(2.0f * x) + 1.0f);
}

// ---------------------------------------------------------------------------
// fp32 -> bf16 bulk convert (weights / z), vectorized 4-wide
// ---------------------------------------------------------------------------
__global__ void cvt_f32_bf16(const float* __restrict__ in, __bf16* __restrict__ out,
                             int n4) {
  int i = blockIdx.x * blockDim.x + threadIdx.x;
  if (i >= n4) return;
  float4 v = reinterpret_cast<const float4*>(in)[i];
  v4bf o;
  o.x = (__bf16)v.x; o.y = (__bf16)v.y; o.z = (__bf16)v.z; o.w = (__bf16)v.w;
  reinterpret_cast<v4bf*>(out)[i] = o;
}

// zero c state + broadcast init_emb into bf16 x0 buffer
__global__ void init_state_kernel(float* __restrict__ c, __bf16* __restrict__ xbuf,
                                  const float* __restrict__ emb) {
  int i = blockIdx.x * blockDim.x + threadIdx.x;
  if (i < LNUM * BSZ * HDIM) c[i] = 0.0f;
  if (i < BSZ * HDIM) xbuf[i] = (__bf16)emb[i & (HDIM - 1)];
}

// ---------------------------------------------------------------------------
// h0 = tanh(z @ to_h_W^T + to_h_b), scattered to h[L][B][H] (bf16)
// M=512, N=2048, K=1024. Block: 256 thr = 8 waves, tile M=64 x N=64.
// Wave w: m_off=(w&1)*32, n_off=(w>>1)*16 -> 2 WMMA tiles per wave.
// ---------------------------------------------------------------------------
__global__ __launch_bounds__(256) void init_h_kernel(
    const __bf16* __restrict__ z, const __bf16* __restrict__ Wt,
    const float* __restrict__ bias, __bf16* __restrict__ h) {
  const int lane = threadIdx.x & 31;
  const int wave = threadIdx.x >> 5;
  const int m0 = blockIdx.y * 64 + (wave & 1) * 32;
  const int n0 = blockIdx.x * 64 + (wave >> 1) * 16;
  const int mr = lane & 15;

  const __bf16* arow0 = z + (size_t)(m0 + mr) * HDIM;
  const __bf16* arow1 = z + (size_t)(m0 + 16 + mr) * HDIM;
  const __bf16* wrow = Wt + (size_t)(n0 + mr) * HDIM;

  v8f acc0 = {0.f, 0.f, 0.f, 0.f, 0.f, 0.f, 0.f, 0.f};
  v8f acc1 = acc0;

  for (int k = 0; k < HDIM; k += 32) {
    v16bf a0 = load_a_frag(arow0 + k, lane);
    v16bf a1 = load_a_frag(arow1 + k, lane);
    v16bf b = load_b_frag(wrow + k, lane);
    acc0 = wmma_bf16(a0, b, acc0);
    acc1 = wmma_bf16(a1, b, acc1);
  }

  const int n = n0 + mr;            // 0..2047
  const float bn = bias[n];
  const int l = n >> 10;            // layer
  const int jj = n & (HDIM - 1);
  const int rbase = (lane >> 4) * 8;
#pragma unroll
  for (int mt = 0; mt < 2; ++mt) {
    v8f a = mt ? acc1 : acc0;
#pragma unroll
    for (int r = 0; r < 8; ++r) {
      const int row = m0 + mt * 16 + rbase + r;
      float v = tanh_f(a[r] + bn);
      h[((size_t)l * BSZ + row) * HDIM + jj] = (__bf16)v;
    }
  }
}

// ---------------------------------------------------------------------------
// Fused LSTM layer step:
//   gates = [x|h] @ [W_ih|W_hh]^T (+biases), then cell update in-register.
// Block: 64 thr = 2 waves. Wave w: jj_off = w*16, M = 64 rows (4 m-tiles).
// Each wave: 16 accumulators (4 gates x 4 m-tiles) over a 64x16 (b,jj) patch.
// Weight fragments are shared across 4 m-tiles -> load:math = 16 b128 : 16 wmma.
// Grid: (H/32, B/64) = (32, 8) = 256 blocks.
// ---------------------------------------------------------------------------
__global__ __launch_bounds__(64) void lstm_cell_kernel(
    const __bf16* __restrict__ x, const __bf16* __restrict__ hprev,
    const __bf16* __restrict__ Wih, const __bf16* __restrict__ Whh,
    const float* __restrict__ bih, const float* __restrict__ bhh,
    float* __restrict__ cstate, __bf16* __restrict__ hout,
    float* __restrict__ yout) {
  const int lane = threadIdx.x & 31;
  const int wave = threadIdx.x >> 5;
  const int m0 = blockIdx.y * 64;
  const int jj0 = blockIdx.x * 32 + wave * 16;
  const int mr = lane & 15;

  const __bf16* xrow[4];
  const __bf16* hrow[4];
#pragma unroll
  for (int mt = 0; mt < 4; ++mt) {
    xrow[mt] = x + (size_t)(m0 + mt * 16 + mr) * HDIM;
    hrow[mt] = hprev + (size_t)(m0 + mt * 16 + mr) * HDIM;
  }

  const __bf16* wih[4];
  const __bf16* whh[4];
#pragma unroll
  for (int g = 0; g < 4; ++g) {
    wih[g] = Wih + (size_t)(g * HDIM + jj0 + mr) * HDIM;
    whh[g] = Whh + (size_t)(g * HDIM + jj0 + mr) * HDIM;
  }

  v8f zero8 = {0.f, 0.f, 0.f, 0.f, 0.f, 0.f, 0.f, 0.f};
  v8f acc[4][4];
#pragma unroll
  for (int g = 0; g < 4; ++g)
#pragma unroll
    for (int mt = 0; mt < 4; ++mt) acc[g][mt] = zero8;

  // Phase 1: x @ W_ih^T  (unroll 2 so the scheduler can overlap two
  // iterations' b128 clauses against 32 in-flight WMMAs)
#pragma unroll 2
  for (int k = 0; k < HDIM; k += 32) {
    v16bf a0 = load_a_frag(xrow[0] + k, lane);
    v16bf a1 = load_a_frag(xrow[1] + k, lane);
    v16bf a2 = load_a_frag(xrow[2] + k, lane);
    v16bf a3 = load_a_frag(xrow[3] + k, lane);
#pragma unroll
    for (int g = 0; g < 4; ++g) {
      v16bf b = load_b_frag(wih[g] + k, lane);
      acc[g][0] = wmma_bf16(a0, b, acc[g][0]);
      acc[g][1] = wmma_bf16(a1, b, acc[g][1]);
      acc[g][2] = wmma_bf16(a2, b, acc[g][2]);
      acc[g][3] = wmma_bf16(a3, b, acc[g][3]);
    }
  }
  // Phase 2: h_prev @ W_hh^T
#pragma unroll 2
  for (int k = 0; k < HDIM; k += 32) {
    v16bf a0 = load_a_frag(hrow[0] + k, lane);
    v16bf a1 = load_a_frag(hrow[1] + k, lane);
    v16bf a2 = load_a_frag(hrow[2] + k, lane);
    v16bf a3 = load_a_frag(hrow[3] + k, lane);
#pragma unroll
    for (int g = 0; g < 4; ++g) {
      v16bf b = load_b_frag(whh[g] + k, lane);
      acc[g][0] = wmma_bf16(a0, b, acc[g][0]);
      acc[g][1] = wmma_bf16(a1, b, acc[g][1]);
      acc[g][2] = wmma_bf16(a2, b, acc[g][2]);
      acc[g][3] = wmma_bf16(a3, b, acc[g][3]);
    }
  }

  // Cell update (PyTorch gate order i, f, g, o)
  const int jj = jj0 + mr;
  const float bi = bih[0 * HDIM + jj] + bhh[0 * HDIM + jj];
  const float bf = bih[1 * HDIM + jj] + bhh[1 * HDIM + jj];
  const float bg = bih[2 * HDIM + jj] + bhh[2 * HDIM + jj];
  const float bo = bih[3 * HDIM + jj] + bhh[3 * HDIM + jj];
  const int rbase = (lane >> 4) * 8;
#pragma unroll
  for (int mt = 0; mt < 4; ++mt) {
#pragma unroll
    for (int r = 0; r < 8; ++r) {
      const int row = m0 + mt * 16 + rbase + r;
      const size_t idx = (size_t)row * HDIM + jj;
      float gi = sigm_f(acc[0][mt][r] + bi);
      float gf = sigm_f(acc[1][mt][r] + bf);
      float gg = tanh_f(acc[2][mt][r] + bg);
      float go = sigm_f(acc[3][mt][r] + bo);
      float cn = gf * cstate[idx] + gi * gg;
      cstate[idx] = cn;
      float hn = go * tanh_f(cn);
      hout[idx] = (__bf16)hn;
      if (yout) __builtin_nontemporal_store(hn, &yout[idx]);
    }
  }
}

// ---------------------------------------------------------------------------
// Host driver
// ---------------------------------------------------------------------------
extern "C" void kernel_launch(void* const* d_in, const int* in_sizes, int n_in,
                              void* d_out, int out_size, void* d_ws, size_t ws_size,
                              hipStream_t stream) {
  const float* z    = (const float*)d_in[0];
  const float* tohW = (const float*)d_in[1];
  const float* tohB = (const float*)d_in[2];
  const float* emb  = (const float*)d_in[3];
  const float* Wih  = (const float*)d_in[4];
  const float* Whh  = (const float*)d_in[5];
  const float* bih  = (const float*)d_in[6];
  const float* bhh  = (const float*)d_in[7];
  float* out = (float*)d_out;

  const size_t WL   = (size_t)LNUM * GDIM * HDIM;  // per weight tensor elems
  const size_t TOHW = (size_t)LNUM * HDIM * HDIM;
  const size_t ZN   = (size_t)BSZ * HDIM;
  const size_t HN   = (size_t)LNUM * BSZ * HDIM;

  // Workspace layout (bf16 unless noted), ~48 MB total
  __bf16* Wih_bf  = (__bf16*)d_ws;
  __bf16* Whh_bf  = Wih_bf + WL;
  __bf16* tohW_bf = Whh_bf + WL;
  __bf16* z_bf    = tohW_bf + TOHW;
  __bf16* hA      = z_bf + ZN;       // [L][B][H]
  __bf16* hB      = hA + HN;         // [L][B][H]
  __bf16* xbuf    = hB + HN;         // [B][H] x0
  float*  cst     = (float*)(xbuf + ZN);  // [L][B][H] fp32

  // 1) bf16 conversions (deterministic, every launch)
  {
    int n4 = (int)(WL / 4);
    cvt_f32_bf16<<<(n4 + 255) / 256, 256, 0, stream>>>(Wih, Wih_bf, n4);
    cvt_f32_bf16<<<(n4 + 255) / 256, 256, 0, stream>>>(Whh, Whh_bf, n4);
    int n4w = (int)(TOHW / 4);
    cvt_f32_bf16<<<(n4w + 255) / 256, 256, 0, stream>>>(tohW, tohW_bf, n4w);
    int n4z = (int)(ZN / 4);
    cvt_f32_bf16<<<(n4z + 255) / 256, 256, 0, stream>>>(z, z_bf, n4z);
  }

  // 2) zero c, broadcast x0
  init_state_kernel<<<(int)((HN + 255) / 256), 256, 0, stream>>>(cst, xbuf, emb);

  // 3) h0 = tanh(z @ to_h_W^T + b) -> hA
  init_h_kernel<<<dim3(LNUM * HDIM / 64, BSZ / 64), 256, 0, stream>>>(
      z_bf, tohW_bf, tohB, hA);

  // 4) T time steps, 2 layers each, h double-buffered
  __bf16* hp = hA;
  __bf16* hc = hB;
  for (int t = 0; t < TSTEPS; ++t) {
    const __bf16* xin = (t == 0) ? (const __bf16*)xbuf
                                 : (const __bf16*)(hp + (size_t)BSZ * HDIM);
    // layer 0
    lstm_cell_kernel<<<dim3(HDIM / 32, BSZ / 64), 64, 0, stream>>>(
        xin, hp, Wih_bf, Whh_bf, bih, bhh, cst, hc, (float*)nullptr);
    // layer 1 (input = layer0's fresh h; also emits y_t fp32)
    lstm_cell_kernel<<<dim3(HDIM / 32, BSZ / 64), 64, 0, stream>>>(
        (const __bf16*)hc, hp + (size_t)BSZ * HDIM,
        Wih_bf + (size_t)GDIM * HDIM, Whh_bf + (size_t)GDIM * HDIM,
        bih + GDIM, bhh + GDIM,
        cst + (size_t)BSZ * HDIM, hc + (size_t)BSZ * HDIM,
        out + (size_t)t * BSZ * HDIM);
    __bf16* tmp = hp; hp = hc; hc = tmp;
  }
}
